// SpatialTransformerBlock_48120813584466
// MI455X (gfx1250) — matrix operand order repaired and checked by reading, not compiled, not run on hardware
//
#include <hip/hip_runtime.h>
#include <hip/hip_bf16.h>
#include <math.h>

// ---------------------------------------------------------------------------
// SpatialTransformerBlock on MI455X (gfx1250, wave32, WMMA).
// All GEMMs run on v_wmma_f32_16x16x32_bf16 (f32 accumulate); global->LDS
// staging uses GLOBAL_LOAD_ASYNC_TO_LDS_B128 when the toolchain exposes it.
// Shapes: B=16, N=1024 (32x32), D=384, H1=6 (hd 64), H2=3 (hd 128), MLP=1536.
// ---------------------------------------------------------------------------

typedef __bf16 bf16;
typedef __attribute__((ext_vector_type(16))) __bf16 bf16x16;
typedef __attribute__((ext_vector_type(8)))  float  floatx8;
typedef __attribute__((ext_vector_type(4)))  int    i32x4;

union Frag16 { bf16x16 v; unsigned int u[8]; };

__device__ __forceinline__ float bf2f(bf16 x) { return (float)x; }
__device__ __forceinline__ bf16  f2bf(float x) { return (bf16)x; }

__device__ __forceinline__ floatx8 fzero8() {
  floatx8 z;
#pragma unroll
  for (int i = 0; i < 8; ++i) z[i] = 0.0f;
  return z;
}

// --- async global->LDS copy (CDNA5 GLOBAL_LOAD_ASYNC_TO_LDS_B128) ----------
#if defined(__gfx1250__) && __has_builtin(__builtin_amdgcn_global_load_async_to_lds_b128)
#define HAVE_ASYNC_LDS 1
#else
#define HAVE_ASYNC_LDS 0
#endif

__device__ __forceinline__ void async_copy16(const bf16* g, bf16* l) {
#if HAVE_ASYNC_LDS
  // Builtin expects (i32x4 AS(1)* gsrc, i32x4 AS(3)* ldst, imm offset, imm cpol)
  bf16* gnc = const_cast<bf16*>(g);
  __builtin_amdgcn_global_load_async_to_lds_b128(
      (__attribute__((address_space(1))) i32x4*)gnc,
      (__attribute__((address_space(3))) i32x4*)l, 0, 0);
#else
  *(uint4*)l = *(const uint4*)g;
#endif
}

__device__ __forceinline__ void wait_async0() {
#if HAVE_ASYNC_LDS
#if __has_builtin(__builtin_amdgcn_s_wait_asynccnt)
  __builtin_amdgcn_s_wait_asynccnt(0);
#else
  asm volatile("s_wait_asynccnt 0x0" ::: "memory");
#endif
#endif
}

// WMMA fragment layouts (wave32, 16x16x32 bf16; CDNA5 ISA §7.12.2):
//   A (16x32): lane L: M = L%16, half = L/16; dword v holds K pair
//              kp = (v<4 ? 2v : 16+2(v-4)) + 8*half
//   B (32x16): lane L: N = L%16, half = L/16; hw order as A; this code loads A and B with kp = 16*half + 2v
//   C/D (16x16 f32): lane L: N = L%16; vgpr v: M = v + 8*(L/16)
__device__ __forceinline__ int a_kpair(int v, int half) {
  return (v < 4 ? 2 * v : 16 + 2 * (v - 4)) + 8 * half;
}

// ---------------------------------------------------------------------------
// Generic bf16 WMMA GEMM:  C[M,N] = A[M,K] * W[N,K]^T + bias
//   A row-major bf16, W row-major [N,K] bf16 (torch "out,in" weight layout, so
//   B[k,n] = W[n,k] and K-pairs are contiguous in memory).
// Block: 128 threads (4 waves, 2x2), block tile 64x128, wave tile 32x64,
// K step 32, LDS tiles filled with async-to-LDS DMA. Dims divide tiles exactly.
// EPI: 0 = bf16 out, 1 = bf16 out + exact GELU, 2 = f32 out = resid + scale*(acc+bias)
// ---------------------------------------------------------------------------
template <int EPI>
__global__ __launch_bounds__(128) void gemm_bf16_wmma(
    const bf16* __restrict__ A, int lda, const bf16* __restrict__ W, int ldw,
    const float* __restrict__ bias, int M, int N, int K,
    bf16* __restrict__ outB, float* __restrict__ outF,
    const float* __restrict__ resid, float scale) {
  __shared__ bf16 lA[64 * 32];    // 4 KB
  __shared__ bf16 lB[128 * 32];   // 8 KB

  const int tid  = threadIdx.x;
  const int wave = tid >> 5, lane = tid & 31;
  const int half = lane >> 4, l16 = lane & 15;
  const int mw = wave >> 1, nw = wave & 1;
  const int blockM = blockIdx.y * 64;
  const int blockN = blockIdx.x * 128;
  const int waveM = blockM + mw * 32;
  const int waveN = blockN + nw * 64;

  floatx8 acc[2][4];
#pragma unroll
  for (int mi = 0; mi < 2; ++mi)
#pragma unroll
    for (int ni = 0; ni < 4; ++ni) acc[mi][ni] = fzero8();

  for (int k0 = 0; k0 < K; k0 += 32) {
    // Stage A tile 64x32 (thread -> 32B) and B tile 128x32 (thread -> 64B)
    // via async DMA straight into LDS (no VGPR round-trip).
    {
      int row = tid >> 1, seg = tid & 1;
      const bf16* srcA = A + (size_t)(blockM + row) * lda + k0 + seg * 16;
      bf16* dstA = &lA[row * 32 + seg * 16];
      async_copy16(srcA, dstA);
      async_copy16(srcA + 8, dstA + 8);
      const bf16* srcB = W + (size_t)(blockN + tid) * ldw + k0;
      bf16* dstB = &lB[tid * 32];
      async_copy16(srcB, dstB);
      async_copy16(srcB + 8, dstB + 8);
      async_copy16(srcB + 16, dstB + 16);
      async_copy16(srcB + 24, dstB + 24);
    }
    if (k0 + 32 < K) {  // hint next slabs toward the caches
      __builtin_prefetch(A + (size_t)(blockM + (tid >> 1)) * lda + k0 + 32, 0, 1);
      __builtin_prefetch(W + (size_t)(blockN + tid) * ldw + k0 + 32, 0, 1);
    }
    wait_async0();
    __syncthreads();

    Frag16 a[2], b[4];
#pragma unroll
    for (int mi = 0; mi < 2; ++mi) {
      const bf16* base = &lA[(mw * 32 + mi * 16 + l16) * 32];
#pragma unroll
      for (int v = 0; v < 8; ++v)
        a[mi].u[v] = *(const unsigned int*)(base + 16 * half + 2 * v);
    }
#pragma unroll
    for (int ni = 0; ni < 4; ++ni) {
      const bf16* base = &lB[(nw * 64 + ni * 16 + l16) * 32];
#pragma unroll
      for (int v = 0; v < 8; ++v)
        b[ni].u[v] = *(const unsigned int*)(base + 16 * half + 2 * v);
    }
#pragma unroll
    for (int mi = 0; mi < 2; ++mi)
#pragma unroll
      for (int ni = 0; ni < 4; ++ni)
        acc[mi][ni] = __builtin_amdgcn_wmma_f32_16x16x32_bf16(
            false, a[mi].v, false, b[ni].v, (short)0, acc[mi][ni], false, false);
    __syncthreads();
  }

#pragma unroll
  for (int mi = 0; mi < 2; ++mi)
#pragma unroll
    for (int ni = 0; ni < 4; ++ni) {
      int n = waveN + ni * 16 + l16;
      float bv = bias ? bias[n] : 0.0f;
#pragma unroll
      for (int v = 0; v < 8; ++v) {
        int m = waveM + mi * 16 + v + 8 * half;
        float val = acc[mi][ni][v] + bv;
        size_t idx = (size_t)m * N + n;
        if (EPI == 0) {
          outB[idx] = f2bf(val);
        } else if (EPI == 1) {  // exact GELU
          outB[idx] = f2bf(0.5f * val * (1.0f + erff(val * 0.70710678118654752f)));
        } else {
          outF[idx] = resid[idx] + scale * val;
        }
      }
    }
}

// ---------------------------------------------------------------------------
// Flash attention (wave32 WMMA). 4 waves/block, each owning 16 query rows of
// the SAME (b, head) -> K/V tiles (32 keys x HD) are staged once per step into
// LDS by the whole block (async DMA) and consumed by all 4 waves.
// Online softmax; P re-laid out via a per-wave LDS tile (C->A transpose).
// ---------------------------------------------------------------------------
template <int HD>
__global__ __launch_bounds__(128) void flash_attn_wmma(
    const bf16* __restrict__ Qb, int ldq, const bf16* __restrict__ Kb, int ldk,
    const bf16* __restrict__ Vb, int ldv, bf16* __restrict__ Ob, int ldo,
    int nheads, int seq, float scale) {
  constexpr int KF = HD / 32;        // Q fragments / score K-steps
  constexpr int OF = HD / 16;        // output column fragments
  constexpr int CH = HD / 8;         // 16B chunks per K/V row
  constexpr int PER = (32 * CH) / 128;  // chunks per thread per matrix
  __shared__ bf16 lK[32 * HD];
  __shared__ bf16 lV[32 * HD];
  __shared__ bf16 lP[4][16 * 32];

  const int tid  = threadIdx.x;
  const int wave = tid >> 5, lane = tid & 31;
  const int half = lane >> 4, l16 = lane & 15;

  const int tiles = seq / 64;
  const int bh = blockIdx.x / tiles;
  const int qt = blockIdx.x % tiles;
  const int b = bh / nheads, h = bh % nheads;
  const int q0 = qt * 64 + wave * 16;
  const size_t rowbase = (size_t)b * seq;
  const int hq = h * HD;

  // Load Q fragments once (A layout; K-pairs contiguous along head dim).
  Frag16 qf[KF];
#pragma unroll
  for (int kk = 0; kk < KF; ++kk) {
    const bf16* src = Qb + (rowbase + q0 + l16) * ldq + hq + kk * 32;
#pragma unroll
    for (int v = 0; v < 8; ++v)
      qf[kk].u[v] = *(const unsigned int*)(src + 16 * half + 2 * v);
  }

  floatx8 o[OF];
#pragma unroll
  for (int f = 0; f < OF; ++f) o[f] = fzero8();
  float mrow[8], lrow[8];
#pragma unroll
  for (int v = 0; v < 8; ++v) { mrow[v] = -1e30f; lrow[v] = 0.0f; }

  bf16* lp = lP[wave];

  for (int j0 = 0; j0 < seq; j0 += 32) {
    // Cooperative async staging of this 32-key K and V tile into LDS.
#pragma unroll
    for (int c = 0; c < PER; ++c) {
      int chunk = tid + c * 128;
      int row = chunk / CH, col = (chunk % CH) * 8;
      async_copy16(Kb + (rowbase + j0 + row) * ldk + hq + col, &lK[row * HD + col]);
      async_copy16(Vb + (rowbase + j0 + row) * ldv + hq + col, &lV[row * HD + col]);
    }
    wait_async0();
    __syncthreads();

    // S(16x32) = Q * K^T for two 16-key tiles (B[k=d, n=key]; d-pairs contiguous)
    floatx8 s[2];
    s[0] = fzero8(); s[1] = fzero8();
#pragma unroll
    for (int jj = 0; jj < 2; ++jj) {
#pragma unroll
      for (int kk = 0; kk < KF; ++kk) {
        Frag16 kf;
        const bf16* src = &lK[(jj * 16 + l16) * HD + kk * 32];
#pragma unroll
        for (int v = 0; v < 8; ++v)
          kf.u[v] = *(const unsigned int*)(src + 16 * half + 2 * v);
        s[jj] = __builtin_amdgcn_wmma_f32_16x16x32_bf16(
            false, qf[kk].v, false, kf.v, (short)0, s[jj], false, false);
      }
    }

    // Online softmax across the 32 keys. Row v+8*half lives across the 16-lane
    // half in lanes -> butterfly shfl (masks 1,2,4,8) reduces within the half.
    float p0s[8], p1s[8];
#pragma unroll
    for (int v = 0; v < 8; ++v) {
      float a0 = s[0][v] * scale, a1 = s[1][v] * scale;
      float mx = fmaxf(a0, a1);
      mx = fmaxf(mx, __shfl_xor(mx, 1, 32));
      mx = fmaxf(mx, __shfl_xor(mx, 2, 32));
      mx = fmaxf(mx, __shfl_xor(mx, 4, 32));
      mx = fmaxf(mx, __shfl_xor(mx, 8, 32));
      float nm = fmaxf(mrow[v], mx);
      float alpha = __expf(mrow[v] - nm);
      mrow[v] = nm;
      float p0 = __expf(a0 - nm), p1 = __expf(a1 - nm);
      float rs = p0 + p1;
      rs += __shfl_xor(rs, 1, 32);
      rs += __shfl_xor(rs, 2, 32);
      rs += __shfl_xor(rs, 4, 32);
      rs += __shfl_xor(rs, 8, 32);
      lrow[v] = lrow[v] * alpha + rs;
#pragma unroll
      for (int f = 0; f < OF; ++f) o[f][v] *= alpha;
      p0s[v] = p0; p1s[v] = p1;
    }

    // C-layout P -> private LDS tile (16 rows x 32 keys) -> A-layout fragment.
#pragma unroll
    for (int v = 0; v < 8; ++v) {
      int row = v + 8 * half;
      lp[row * 32 + l16]      = f2bf(p0s[v]);
      lp[row * 32 + 16 + l16] = f2bf(p1s[v]);
    }
    Frag16 pf;
#pragma unroll
    for (int v = 0; v < 8; ++v)
      pf.u[v] = *(const unsigned int*)(lp + l16 * 32 + 16 * half + 2 * v);

    // O += P * V from the staged LDS tile. B-layout needs key-pairs per dword:
    // two bf16 LDS reads from consecutive V rows packed into one dword.
#pragma unroll
    for (int f = 0; f < OF; ++f) {
      Frag16 vf;
#pragma unroll
      for (int v = 0; v < 8; ++v) {
        int jl = 16 * half + 2 * v;
        const bf16* p = &lV[jl * HD + f * 16 + l16];
        unsigned int lo = *(const unsigned short*)p;
        unsigned int hi = *(const unsigned short*)(p + HD);
        vf.u[v] = lo | (hi << 16);
      }
      o[f] = __builtin_amdgcn_wmma_f32_16x16x32_bf16(
          false, pf.v, false, vf.v, (short)0, o[f], false, false);
    }
    __syncthreads();  // all waves done with lK/lV before next staging round
  }

  // Normalize and store (head-concatenated bf16 output).
#pragma unroll
  for (int v = 0; v < 8; ++v) {
    float inv = 1.0f / lrow[v];
    int m = q0 + v + 8 * half;
#pragma unroll
    for (int f = 0; f < OF; ++f)
      Ob[(rowbase + m) * ldo + hq + f * 16 + l16] = f2bf(o[f][v] * inv);
  }
}

// ---------------------------------------------------------------------------
// LayerNorm: one wave per 384-elem row, bf16 output for WMMA consumption.
// ---------------------------------------------------------------------------
__global__ __launch_bounds__(256) void layernorm_to_bf16(
    const float* __restrict__ x, const float* __restrict__ g,
    const float* __restrict__ bsh, bf16* __restrict__ out, int D) {
  int row = blockIdx.x * 8 + (threadIdx.x >> 5);
  int lane = threadIdx.x & 31;
  const float* xr = x + (size_t)row * D;
  float s = 0.0f, ss = 0.0f;
  for (int i = lane; i < D; i += 32) { float v = xr[i]; s += v; ss += v * v; }
#pragma unroll
  for (int m = 1; m < 32; m <<= 1) {
    s += __shfl_xor(s, m, 32);
    ss += __shfl_xor(ss, m, 32);
  }
  float mean = s / (float)D;
  float var = ss / (float)D - mean * mean;
  float r = rsqrtf(var + 1e-5f);
  bf16* orow = out + (size_t)row * D;
  for (int i = lane; i < D; i += 32)
    orow[i] = f2bf((xr[i] - mean) * r * g[i] + bsh[i]);
}

// 3x3 weighted stencil on the 32x32 grid (center/ortho 1, diag 0.5, normed).
__global__ void spatial_neighbors_bf16(const bf16* __restrict__ xs,
                                       bf16* __restrict__ nb) {
  size_t idx = (size_t)blockIdx.x * blockDim.x + threadIdx.x;
  const size_t total = (size_t)16 * 1024 * 384;
  if (idx >= total) return;
  int d = (int)(idx % 384);
  int n = (int)((idx / 384) % 1024);
  size_t b = idx / ((size_t)384 * 1024);
  int i = n >> 5, j = n & 31;
  float acc = 0.0f, cnt = 0.0f;
#pragma unroll
  for (int di = -1; di <= 1; ++di)
#pragma unroll
    for (int dj = -1; dj <= 1; ++dj) {
      int ii = i + di, jj = j + dj;
      if (ii < 0 || ii > 31 || jj < 0 || jj > 31) continue;
      float w = (di == 0 || dj == 0) ? 1.0f : 0.5f;
      acc += w * bf2f(xs[(b * 1024 + (size_t)(ii * 32 + jj)) * 384 + d]);
      cnt += w;
    }
  nb[idx] = f2bf(acc / cnt);
}

__global__ void cvt_f32_to_bf16(const float* __restrict__ in,
                                bf16* __restrict__ out, int n) {
  int i = blockIdx.x * blockDim.x + threadIdx.x;
  if (i < n) out[i] = f2bf(in[i]);
}

// ---------------------------------------------------------------------------
// Host orchestration
// ---------------------------------------------------------------------------
extern "C" void kernel_launch(void* const* d_in, const int* in_sizes, int n_in,
                              void* d_out, int out_size, void* d_ws, size_t ws_size,
                              hipStream_t stream) {
  (void)in_sizes; (void)n_in; (void)out_size; (void)ws_size;
  const int Bb = 16, N = 1024, D = 384, MLPH = 1536;
  const size_t R = (size_t)Bb * N;  // 16384 rows

  const float* x       = (const float*)d_in[0];
  const float* ln1_g   = (const float*)d_in[1];
  const float* ln1_b   = (const float*)d_in[2];
  const float* attn_wi = (const float*)d_in[3];
  const float* attn_bi = (const float*)d_in[4];
  const float* attn_wo = (const float*)d_in[5];
  const float* attn_bo = (const float*)d_in[6];
  const float* sln_g   = (const float*)d_in[7];
  const float* sln_b   = (const float*)d_in[8];
  const float* sattn_wi = (const float*)d_in[9];
  const float* sattn_bi = (const float*)d_in[10];
  const float* sattn_wo = (const float*)d_in[11];
  const float* sattn_bo = (const float*)d_in[12];
  const float* ln2_g   = (const float*)d_in[13];
  const float* ln2_b   = (const float*)d_in[14];
  const float* mlp_w1  = (const float*)d_in[15];
  const float* mlp_b1  = (const float*)d_in[16];
  const float* mlp_w2  = (const float*)d_in[17];
  const float* mlp_b2  = (const float*)d_in[18];

  // Workspace layout (~168 MB total).
  char* ws = (char*)d_ws;
  size_t off = 0;
  auto alloc = [&](size_t bytes) { size_t o = off; off += (bytes + 255) & ~(size_t)255; return o; };
  float* x_f32 = (float*)(ws + alloc(R * D * 4));
  bf16* xn_bf  = (bf16*)(ws + alloc(R * D * 2));          // LN1 out, reused for LN2 out
  bf16* qkv_bf = (bf16*)(ws + alloc(R * 1152 * 2));       // reused as qs | kvs
  bf16* o_bf   = (bf16*)(ws + alloc(R * D * 2));          // attn outputs
  bf16* xs_bf  = (bf16*)(ws + alloc(R * D * 2));
  bf16* nb_bf  = (bf16*)(ws + alloc(R * D * 2));
  bf16* h_bf   = (bf16*)(ws + alloc(R * MLPH * 2));
  bf16* wi1_bf = (bf16*)(ws + alloc((size_t)1152 * D * 2));
  bf16* wo1_bf = (bf16*)(ws + alloc((size_t)D * D * 2));
  bf16* wi2_bf = (bf16*)(ws + alloc((size_t)1152 * D * 2));
  bf16* wo2_bf = (bf16*)(ws + alloc((size_t)D * D * 2));
  bf16* w1_bf  = (bf16*)(ws + alloc((size_t)MLPH * D * 2));
  bf16* w2_bf  = (bf16*)(ws + alloc((size_t)D * MLPH * 2));
  bf16* qs_bf  = qkv_bf;                 // [R,384]
  bf16* kvs_bf = qkv_bf + R * D;         // [R,768]

  // Convert weights to bf16 (inputs may change every call; no caching).
  auto cvt = [&](const float* src, bf16* dst, int n) {
    cvt_f32_to_bf16<<<(n + 255) / 256, 256, 0, stream>>>(src, dst, n);
  };
  cvt(attn_wi, wi1_bf, 1152 * D);
  cvt(attn_wo, wo1_bf, D * D);
  cvt(sattn_wi, wi2_bf, 1152 * D);
  cvt(sattn_wo, wo2_bf, D * D);
  cvt(mlp_w1, w1_bf, MLPH * D);
  cvt(mlp_w2, w2_bf, D * MLPH);

  // Residual stream (f32) starts as a copy of x.
  (void)hipMemcpyAsync(x_f32, x, R * D * 4, hipMemcpyDeviceToDevice, stream);

  dim3 blk(128);
  // ---- Stage 1: LN1 -> QKV -> attention(6 heads, hd 64) -> out-proj + resid
  layernorm_to_bf16<<<R / 8, 256, 0, stream>>>(x_f32, ln1_g, ln1_b, xn_bf, D);
  gemm_bf16_wmma<0><<<dim3(1152 / 128, R / 64), blk, 0, stream>>>(
      xn_bf, D, wi1_bf, D, attn_bi, (int)R, 1152, D, qkv_bf, nullptr, nullptr, 0.f);
  flash_attn_wmma<64><<<16 * 6 * (N / 64), blk, 0, stream>>>(
      qkv_bf, 1152, qkv_bf + 384, 1152, qkv_bf + 768, 1152, o_bf, D, 6, N, 0.125f);
  gemm_bf16_wmma<2><<<dim3(D / 128, R / 64), blk, 0, stream>>>(
      o_bf, D, wo1_bf, D, attn_bo, (int)R, D, D, nullptr, x_f32, x_f32, 1.0f);

  // ---- Stage 2: sLN -> neighbors -> Q/KV proj -> attention(3 heads, hd 128)
  layernorm_to_bf16<<<R / 8, 256, 0, stream>>>(x_f32, sln_g, sln_b, xs_bf, D);
  {
    size_t total = R * D;
    spatial_neighbors_bf16<<<(unsigned)((total + 255) / 256), 256, 0, stream>>>(xs_bf, nb_bf);
  }
  gemm_bf16_wmma<0><<<dim3(D / 128, R / 64), blk, 0, stream>>>(
      xs_bf, D, wi2_bf, D, sattn_bi, (int)R, D, D, qs_bf, nullptr, nullptr, 0.f);
  gemm_bf16_wmma<0><<<dim3(768 / 128, R / 64), blk, 0, stream>>>(
      nb_bf, D, wi2_bf + (size_t)D * D, D, sattn_bi + D, (int)R, 768, D,
      kvs_bf, nullptr, nullptr, 0.f);
  flash_attn_wmma<128><<<16 * 3 * (N / 64), blk, 0, stream>>>(
      qs_bf, D, kvs_bf, 768, kvs_bf + 384, 768, o_bf, D, 3, N, 0.0883883476483184f);
  gemm_bf16_wmma<2><<<dim3(D / 128, R / 64), blk, 0, stream>>>(
      o_bf, D, wo2_bf, D, sattn_bo, (int)R, D, D, nullptr, x_f32, x_f32, 0.5f);

  // ---- Stage 3: LN2 -> MLP (GELU) -> out GEMM fused with final residual
  layernorm_to_bf16<<<R / 8, 256, 0, stream>>>(x_f32, ln2_g, ln2_b, xn_bf, D);
  gemm_bf16_wmma<1><<<dim3(MLPH / 128, R / 64), blk, 0, stream>>>(
      xn_bf, D, w1_bf, D, mlp_b1, (int)R, MLPH, D, h_bf, nullptr, nullptr, 0.f);
  gemm_bf16_wmma<2><<<dim3(D / 128, R / 64), blk, 0, stream>>>(
      h_bf, MLPH, w2_bf, MLPH, mlp_b2, (int)R, D, MLPH, nullptr, (float*)d_out,
      x_f32, 1.0f);
}
